// MultiheadAttention_62508954026065
// MI455X (gfx1250) — matrix operand order repaired and checked by reading, not compiled, not run on hardware
//
#include <hip/hip_runtime.h>
#include <hip/hip_bf16.h>
#include <stdint.h>

// ---------------------------------------------------------------------------
// MultiheadAttention for MI455X (gfx1250): bf16 WMMA pipeline with
// async-to-LDS staging (GLOBAL_LOAD_ASYNC_TO_LDS_B128 / ASYNCcnt).
//   1) Q/K/V projections: fp32 x @ W.T + b  -> bf16   (wmma f32_16x16x32_bf16)
//   2) flash-attention:   streaming softmax(QK^T)V    (wmma + v_exp + shfl,
//                         double-buffered async K/V staging)
//   3) out projection:    bf16 attn @ Wo.T + bo -> fp32 (async A staging)
// ---------------------------------------------------------------------------

#define EMBED 768
#define BATCH 4
#define SEQ   2048
#define NHEAD 12
#define HD    64
#define M_TOT (BATCH * SEQ) /* 8192 */

typedef __attribute__((ext_vector_type(16))) __bf16        v16bf;
typedef __attribute__((ext_vector_type(8)))  float         v8f;
typedef __attribute__((ext_vector_type(4)))  float         f32x4;
typedef __attribute__((ext_vector_type(8)))  unsigned short us8;
typedef __attribute__((ext_vector_type(4)))  unsigned short us4;

union Frag {
    v16bf v;
    unsigned short u[16];
};

__device__ __forceinline__ unsigned short f2bf(float f) {
    // round-to-nearest-even fp32 -> bf16
    unsigned int u = __builtin_bit_cast(unsigned int, f);
    u += 0x7FFFu + ((u >> 16) & 1u);
    return (unsigned short)(u >> 16);
}

// ---- CDNA5 async global->LDS copy (16B per lane), ASYNCcnt-tracked --------
// GV mode: vdst holds the LDS byte address, vaddr the 64-bit global address.
// The low 32 bits of a generic pointer into LDS are the LDS byte offset
__device__ __forceinline__ void async_b128(const unsigned short* g, unsigned short* l) {
    unsigned lo = (unsigned)(uintptr_t)l;
    asm volatile("global_load_async_to_lds_b128 %0, %1, off"
                 :: "v"(lo), "v"(g) : "memory");
}

__device__ __forceinline__ void wait_async0() {
    asm volatile("s_wait_asynccnt 0" ::: "memory");
}

// ---------------------------------------------------------------------------
// GEMM: Out[M x 768] = A[M x 768] @ W.T + bias, optional scale.
// Block tile 128x128, K-step 32, 8 waves of 32 (wave32), wave tile 32x64.
// ---------------------------------------------------------------------------
#define BM 128
#define BN 128
#define BK 32
#define LKP 40 /* padded lds K-stride (ushorts) */

template <bool A_BF16, bool OUT_F32>
__global__ __launch_bounds__(256, 2)
void gemm_bias_kernel(const void* __restrict__ Ap,
                      const float* __restrict__ W,
                      const float* __restrict__ bias,
                      void* __restrict__ Outp,
                      float scale)
{
    __shared__ unsigned short sA[BM * LKP];
    __shared__ unsigned short sB[BN * LKP];

    const int tid  = threadIdx.x;
    const int lane = tid & 31;
    const int wave = tid >> 5;
    const int wm   = wave >> 1;  // 0..3  (M direction)
    const int wn   = wave & 1;   // 0..1  (N direction)
    const int hlf  = lane >> 4;  // which 16-lane half
    const int ln15 = lane & 15;

    const int n0 = blockIdx.x * BN;
    const int m0 = blockIdx.y * BM;

    // staging map: 2 threads per row, 16 contiguous k each
    const int lrow = tid >> 1;
    const int lks  = (tid & 1) * 16;

    v8f acc[2][4] = {};

    for (int k0 = 0; k0 < EMBED; k0 += BK) {
        // ---- stage A tile ----
        if (A_BF16) {
            // already bf16 in memory: DMA straight into LDS, no VGPR round-trip
            const unsigned short* A =
                (const unsigned short*)Ap + (size_t)(m0 + lrow) * EMBED + k0 + lks;
            async_b128(A,     &sA[lrow * LKP + lks]);
            async_b128(A + 8, &sA[lrow * LKP + lks + 8]);
        } else {
            const float* A = (const float*)Ap;
            const f32x4* src = (const f32x4*)(A + (size_t)(m0 + lrow) * EMBED + k0 + lks);
#pragma unroll
            for (int j = 0; j < 4; ++j) {
                f32x4 x = src[j];
                us4 hc;
                hc[0] = f2bf(x[0]); hc[1] = f2bf(x[1]);
                hc[2] = f2bf(x[2]); hc[3] = f2bf(x[3]);
                *(us4*)&sA[lrow * LKP + lks + j * 4] = hc;
            }
        }
        // ---- stage W tile rows [n][k] (B matrix needs no transpose for x@W.T)
        {
            const f32x4* src = (const f32x4*)(W + (size_t)(n0 + lrow) * EMBED + k0 + lks);
#pragma unroll
            for (int j = 0; j < 4; ++j) {
                f32x4 x = src[j];
                us4 hc;
                hc[0] = f2bf(x[0]); hc[1] = f2bf(x[1]);
                hc[2] = f2bf(x[2]); hc[3] = f2bf(x[3]);
                *(us4*)&sB[lrow * LKP + lks + j * 4] = hc;
            }
        }
        if (A_BF16) wait_async0();
        __syncthreads();

        // ---- A fragments: 16x32, lane = M row, K split 0..7/8..15 + 16..23/24..31
        Frag af[2];
#pragma unroll
        for (int mt = 0; mt < 2; ++mt) {
            const unsigned short* p = &sA[(wm * 32 + mt * 16 + ln15) * LKP];
            const int ko = hlf ? 8 : 0;
#pragma unroll
            for (int i = 0; i < 8; ++i) {
                af[mt].u[i]     = p[ko + i];
                af[mt].u[8 + i] = p[16 + ko + i];
            }
        }
        // ---- B fragments: 32x16, lane = N col, K split 0..7/8..15 + 16..23/24..31
        Frag bf[4];
#pragma unroll
        for (int nt = 0; nt < 4; ++nt) {
            const unsigned short* p = &sB[(wn * 64 + nt * 16 + ln15) * LKP + (hlf ? 8 : 0)];
#pragma unroll
            for (int i = 0; i < 16; ++i) bf[nt].u[i] = p[i < 8 ? i : i + 8];
        }
#pragma unroll
        for (int mt = 0; mt < 2; ++mt)
#pragma unroll
            for (int nt = 0; nt < 4; ++nt)
                acc[mt][nt] = __builtin_amdgcn_wmma_f32_16x16x32_bf16(
                    false, af[mt].v, false, bf[nt].v, (short)0, acc[mt][nt], false, false);
        __syncthreads();
    }

    // ---- epilogue: C layout lane=N, vgpr r -> M = r + 8*half ----
#pragma unroll
    for (int mt = 0; mt < 2; ++mt) {
#pragma unroll
        for (int nt = 0; nt < 4; ++nt) {
            const int ng = n0 + wn * 64 + nt * 16 + ln15;
            const float bv = bias[ng];
#pragma unroll
            for (int r = 0; r < 8; ++r) {
                const int mg = m0 + wm * 32 + mt * 16 + r + 8 * hlf;
                const float v = (acc[mt][nt][r] + bv) * scale;
                if (OUT_F32)
                    ((float*)Outp)[(size_t)mg * EMBED + ng] = v;
                else
                    ((unsigned short*)Outp)[(size_t)mg * EMBED + ng] = f2bf(v);
            }
        }
    }
}

// ---------------------------------------------------------------------------
// Flash attention: one block per (b, h, 128 queries); wave owns 16 q rows.
// Double-buffered async K/V staging; online softmax; all GEMMs on WMMA.
// ---------------------------------------------------------------------------
#define KCH 32
#define KVP 72 /* lds stride for K/V rows (ushorts) */
#define PPD 40 /* lds stride for P scratch rows */

__global__ __launch_bounds__(256, 2)
void attention_kernel(const unsigned short* __restrict__ Qb,
                      const unsigned short* __restrict__ Kb,
                      const unsigned short* __restrict__ Vb,
                      const unsigned char* __restrict__ kpad,
                      unsigned short* __restrict__ Ob)
{
    __shared__ unsigned short sK[2][KCH * KVP];
    __shared__ unsigned short sV[2][KCH * KVP];
    __shared__ float sMask[2][KCH];
    __shared__ unsigned short sP[8][16 * PPD];

    const int tid  = threadIdx.x;
    const int lane = tid & 31;
    const int wave = tid >> 5;
    const int hlf  = lane >> 4;
    const int ln15 = lane & 15;

    const int b  = blockIdx.z;
    const int h  = blockIdx.y;
    const int q0 = blockIdx.x * 128;
    const int qrow = q0 + wave * 16 + ln15;

    const int key_t  = tid >> 3;       // 0..31: key row this thread stages
    const int seg    = (tid & 7) * 8;  // 8 bf16 = one b128 per thread
    const int ldsoff = key_t * KVP + seg;

    // async prefetch of one 32-key K/V chunk into LDS buffer `buf`
    auto stage = [&](int buf, int kk) {
        const size_t gb = ((size_t)(b * SEQ + kk + key_t)) * EMBED + h * HD + seg;
        async_b128(Kb + gb, &sK[buf][ldsoff]);
        async_b128(Vb + gb, &sV[buf][ldsoff]);
        if (tid < KCH)
            sMask[buf][tid] = kpad[b * SEQ + kk + tid] ? -1e30f : 0.0f;
    };

    // Q fragments resident in registers for the whole pass (16 rows x 64 dims)
    Frag qf[2];
    {
        const unsigned short* qp = Qb + ((size_t)(b * SEQ + qrow)) * EMBED + h * HD;
        const int ko = hlf ? 8 : 0;
#pragma unroll
        for (int kh = 0; kh < 2; ++kh)
#pragma unroll
            for (int i = 0; i < 8; ++i) {
                qf[kh].u[i]     = qp[kh * 32 + ko + i];
                qf[kh].u[8 + i] = qp[kh * 32 + 16 + ko + i];
            }
    }

    float mi[8], li[8];
#pragma unroll
    for (int r = 0; r < 8; ++r) { mi[r] = -1e30f; li[r] = 0.0f; }
    v8f oacc[4] = {};

    stage(0, 0);
    wait_async0();
    __syncthreads();

    int cur = 0;
    for (int kk = 0; kk < SEQ; kk += KCH, cur ^= 1) {
        // overlap: fetch next chunk while computing on the current one
        if (kk + KCH < SEQ)
            stage(cur ^ 1, kk + KCH);

        // ---- scores: S[16 x 32] = Q(16x64) . K^T, two 16x16 C tiles ----
        v8f sc[2] = {};
#pragma unroll
        for (int t = 0; t < 2; ++t) {
#pragma unroll
            for (int kh = 0; kh < 2; ++kh) {
                Frag kf;
                const unsigned short* p =
                    &sK[cur][(t * 16 + ln15) * KVP + kh * 32 + (hlf ? 8 : 0)];
#pragma unroll
                for (int i = 0; i < 16; ++i) kf.u[i] = p[i < 8 ? i : i + 8];
                sc[t] = __builtin_amdgcn_wmma_f32_16x16x32_bf16(
                    false, qf[kh].v, false, kf.v, (short)0, sc[t], false, false);
            }
            const float mv = sMask[cur][t * 16 + ln15];
#pragma unroll
            for (int r = 0; r < 8; ++r) sc[t][r] += mv;
        }

        // ---- online softmax (rows live across a 16-lane half) ----
#pragma unroll
        for (int r = 0; r < 8; ++r) {
            float x = fmaxf(sc[0][r], sc[1][r]);
#pragma unroll
            for (int d = 8; d >= 1; d >>= 1) x = fmaxf(x, __shfl_xor(x, d, 32));
            const float mnew = fmaxf(mi[r], x);
            const float corr = __expf(mi[r] - mnew);
            mi[r] = mnew;
            const float p0 = __expf(sc[0][r] - mnew);
            const float p1 = __expf(sc[1][r] - mnew);
            sc[0][r] = p0;
            sc[1][r] = p1;
            float s = p0 + p1;
#pragma unroll
            for (int d = 8; d >= 1; d >>= 1) s += __shfl_xor(s, d, 32);
            li[r] = li[r] * corr + s;
#pragma unroll
            for (int nt = 0; nt < 4; ++nt) oacc[nt][r] *= corr;
        }

        // ---- C-layout P -> per-wave LDS -> A-layout fragment (bf16) ----
        unsigned short* pw = sP[wave];
#pragma unroll
        for (int t = 0; t < 2; ++t)
#pragma unroll
            for (int r = 0; r < 8; ++r)
                pw[(r + 8 * hlf) * PPD + t * 16 + ln15] = f2bf(sc[t][r]);

        Frag pf;
        {
            const unsigned short* p = &pw[ln15 * PPD];
            const int ko = hlf ? 8 : 0;
#pragma unroll
            for (int i = 0; i < 8; ++i) {
                pf.u[i]     = p[ko + i];
                pf.u[8 + i] = p[16 + ko + i];
            }
        }

        // ---- O += P(16x32) @ V(32x64) ----
#pragma unroll
        for (int nt = 0; nt < 4; ++nt) {
            Frag vf;
            const int kb = hlf ? 8 : 0;
#pragma unroll
            for (int i = 0; i < 16; ++i)
                vf.u[i] = sV[cur][(kb + (i < 8 ? i : i + 8)) * KVP + nt * 16 + ln15];
            oacc[nt] = __builtin_amdgcn_wmma_f32_16x16x32_bf16(
                false, pf.v, false, vf.v, (short)0, oacc[nt], false, false);
        }

        // next buffer's async writes must land before anyone reads it
        wait_async0();
        __syncthreads();
    }

    // ---- epilogue: normalize and scatter heads back to [b, s, h*hd] bf16 ----
#pragma unroll
    for (int nt = 0; nt < 4; ++nt)
#pragma unroll
        for (int r = 0; r < 8; ++r) {
            const int row = r + 8 * hlf;
            const int q   = q0 + wave * 16 + row;
            const float v = oacc[nt][r] / li[r];
            Ob[((size_t)(b * SEQ + q)) * EMBED + h * HD + nt * 16 + ln15] = f2bf(v);
        }
}

// ---------------------------------------------------------------------------
extern "C" void kernel_launch(void* const* d_in, const int* in_sizes, int n_in,
                              void* d_out, int out_size, void* d_ws, size_t ws_size,
                              hipStream_t stream)
{
    const float* query = (const float*)d_in[0];
    const float* key_  = (const float*)d_in[1];
    const float* value = (const float*)d_in[2];
    const unsigned char* mask = (const unsigned char*)d_in[3];
    const float* Wq = (const float*)d_in[4];
    const float* bq = (const float*)d_in[5];
    const float* Wk = (const float*)d_in[6];
    const float* bk = (const float*)d_in[7];
    const float* Wv = (const float*)d_in[8];
    const float* bv = (const float*)d_in[9];
    const float* Wo = (const float*)d_in[10];
    const float* bo = (const float*)d_in[11];
    float* out = (float*)d_out;

    // workspace: 4 bf16 tensors of [8192 x 768] = 4 * 12.6 MB
    const size_t elems = (size_t)M_TOT * EMBED;
    unsigned short* Qb = (unsigned short*)d_ws;
    unsigned short* Kb = Qb + elems;
    unsigned short* Vb = Kb + elems;
    unsigned short* Ab = Vb + elems;

    dim3 blk(256);
    dim3 ggrid(EMBED / BN, M_TOT / BM); // (6, 64)
    const float scaling = 0.125f;       // hd^-0.5 = 1/8

    gemm_bias_kernel<false, false><<<ggrid, blk, 0, stream>>>(query, Wq, bq, Qb, scaling);
    gemm_bias_kernel<false, false><<<ggrid, blk, 0, stream>>>(key_,  Wk, bk, Kb, 1.0f);
    gemm_bias_kernel<false, false><<<ggrid, blk, 0, stream>>>(value, Wv, bv, Vb, 1.0f);

    attention_kernel<<<dim3(SEQ / 128, NHEAD, BATCH), blk, 0, stream>>>(Qb, Kb, Vb, mask, Ab);

    gemm_bias_kernel<true, true><<<ggrid, blk, 0, stream>>>(Ab, Wo, bo, out, 1.0f);
}
